// LoRALinear_13726715478236
// MI455X (gfx1250) — matrix operand order repaired and checked by reading, not compiled, not run on hardware
//
#include <hip/hip_runtime.h>

#define IN_F   4096
#define OUT_F  4096
#define M_TOK  8192
#define RANK   16
#define SCALING 2.0f   // ALPHA / RANK = 32 / 16

typedef __attribute__((ext_vector_type(16))) __bf16 v16bf;
typedef __attribute__((ext_vector_type(8)))  float  v8f;

#define BM   128
#define BN   256
#define KT   32
#define LDT  40        // KT + 8 halves padding (80B row stride, 16B aligned)

__device__ __forceinline__ unsigned short f2bf(float f) {
  unsigned int u = __float_as_uint(f);
  u += 0x7FFFu + ((u >> 16) & 1u);       // round to nearest even
  return (unsigned short)(u >> 16);
}

// ---------------- pass 1: x fp32 -> bf16 ----------------
__global__ void cvt_x_kernel(const float* __restrict__ x,
                             unsigned short* __restrict__ xb) {
  size_t i = ((size_t)blockIdx.x * blockDim.x + threadIdx.x) * 8;
  float4 a = *(const float4*)(x + i);
  float4 b = *(const float4*)(x + i + 4);
  uint4 o;
  o.x = (unsigned)f2bf(a.x) | ((unsigned)f2bf(a.y) << 16);
  o.y = (unsigned)f2bf(a.z) | ((unsigned)f2bf(a.w) << 16);
  o.z = (unsigned)f2bf(b.x) | ((unsigned)f2bf(b.y) << 16);
  o.w = (unsigned)f2bf(b.z) | ((unsigned)f2bf(b.w) << 16);
  *(uint4*)(xb + i) = o;
}

// ------- pass 2: W_eff = W + SCALING * (B @ A), fp32 -> bf16 -------
__global__ void weff_kernel(const float* __restrict__ w,
                            const float* __restrict__ lA,
                            const float* __restrict__ lB,
                            unsigned short* __restrict__ wb) {
  const int o = blockIdx.x;
  float br[RANK];
#pragma unroll
  for (int r = 0; r < RANK; ++r) br[r] = lB[o * RANK + r];

  for (int it = 0; it < IN_F / (256 * 4); ++it) {
    int i = (threadIdx.x + it * 256) * 4;
    float4 wv = *(const float4*)(w + (size_t)o * IN_F + i);
    float d0 = 0.f, d1 = 0.f, d2 = 0.f, d3 = 0.f;
#pragma unroll
    for (int r = 0; r < RANK; ++r) {
      float4 av = *(const float4*)(lA + (size_t)r * IN_F + i);
      d0 += br[r] * av.x; d1 += br[r] * av.y;
      d2 += br[r] * av.z; d3 += br[r] * av.w;
    }
    uint2 p;
    p.x = (unsigned)f2bf(wv.x + SCALING * d0) |
          ((unsigned)f2bf(wv.y + SCALING * d1) << 16);
    p.y = (unsigned)f2bf(wv.z + SCALING * d2) |
          ((unsigned)f2bf(wv.w + SCALING * d3) << 16);
    *(uint2*)(wb + (size_t)o * IN_F + i) = p;
  }
}

// ---------------- pass 3: bf16 WMMA GEMM + bias ----------------
union FragU { uint4 q[2]; v16bf v; };

__global__ __launch_bounds__(256) void lora_gemm_kernel(
    const unsigned short* __restrict__ Xb,    // [M_TOK][IN_F] bf16
    const unsigned short* __restrict__ Wb,    // [OUT_F][IN_F] bf16 (W_eff)
    const float* __restrict__ bias,
    float* __restrict__ out) {                // [M_TOK][OUT_F] fp32
  __shared__ __align__(16) unsigned short ldsA[2][BM * LDT];
  __shared__ __align__(16) unsigned short ldsB[2][BN * LDT];

  const int tid   = threadIdx.x;
  const int lane  = tid & 31;
  const int wave  = tid >> 5;
  const int l15   = lane & 15;
  const int lhalf = lane >> 4;

  const int blockM = blockIdx.y * BM;
  const int blockN = blockIdx.x * BN;
  const int wm0 = (wave & 1) * 64;   // 2 waves along M
  const int wn0 = (wave >> 1) * 64;  // 4 waves along N

  // global-load assignment: A 128x32 halves (32B/thread), B 256x32 (64B/thread)
  const int aRow = tid >> 1;
  const int aCol = (tid & 1) * 16;   // halves
  const unsigned short* aPtr = Xb + (size_t)(blockM + aRow) * IN_F + aCol;
  const unsigned short* bPtr = Wb + (size_t)(blockN + tid) * IN_F;

  v8f c[4][4] = {};
  uint4 aR[2], bR[4];

  // prologue: k-step 0 -> buffer 0
  aR[0] = *(const uint4*)(aPtr);
  aR[1] = *(const uint4*)(aPtr + 8);
  bR[0] = *(const uint4*)(bPtr);
  bR[1] = *(const uint4*)(bPtr + 8);
  bR[2] = *(const uint4*)(bPtr + 16);
  bR[3] = *(const uint4*)(bPtr + 24);
  *(uint4*)(&ldsA[0][aRow * LDT + aCol])     = aR[0];
  *(uint4*)(&ldsA[0][aRow * LDT + aCol + 8]) = aR[1];
  *(uint4*)(&ldsB[0][tid * LDT])             = bR[0];
  *(uint4*)(&ldsB[0][tid * LDT + 8])         = bR[1];
  *(uint4*)(&ldsB[0][tid * LDT + 16])        = bR[2];
  *(uint4*)(&ldsB[0][tid * LDT + 24])        = bR[3];
  __syncthreads();

  const int nK = IN_F / KT;   // 128
  for (int kt = 0; kt < nK; ++kt) {
    const int cur = kt & 1;
    if (kt + 1 < nK) {  // prefetch next K-slab to registers
      const unsigned short* ap = aPtr + (size_t)(kt + 1) * KT;
      const unsigned short* bp = bPtr + (size_t)(kt + 1) * KT;
      aR[0] = *(const uint4*)(ap);
      aR[1] = *(const uint4*)(ap + 8);
      bR[0] = *(const uint4*)(bp);
      bR[1] = *(const uint4*)(bp + 8);
      bR[2] = *(const uint4*)(bp + 16);
      bR[3] = *(const uint4*)(bp + 24);
    }

    // LDS -> fragments (contiguous k = 16*lhalf + s, same order for A and B)
    v16bf af[4], bfr[4];
#pragma unroll
    for (int mi = 0; mi < 4; ++mi) {
      const unsigned short* p =
          &ldsA[cur][(wm0 + mi * 16 + l15) * LDT + lhalf * 16];
      FragU f;
      f.q[0] = *(const uint4*)(p);        // K = koff .. koff+7
      f.q[1] = *(const uint4*)(p + 8);    // K = koff+8 .. koff+15
      af[mi] = f.v;
    }
#pragma unroll
    for (int ni = 0; ni < 4; ++ni) {
      const unsigned short* p =
          &ldsB[cur][(wn0 + ni * 16 + l15) * LDT + lhalf * 16];
      FragU f;
      f.q[0] = *(const uint4*)(p);        // K = koff .. koff+7
      f.q[1] = *(const uint4*)(p + 8);    // K = koff+8 .. koff+15
      bfr[ni] = f.v;
    }

#pragma unroll
    for (int mi = 0; mi < 4; ++mi)
#pragma unroll
      for (int ni = 0; ni < 4; ++ni)
        c[mi][ni] = __builtin_amdgcn_wmma_f32_16x16x32_bf16(
            false, af[mi], false, bfr[ni], (short)0, c[mi][ni], false, false);

    if (kt + 1 < nK) {   // fill the other buffer for the next step
      const int nxt = cur ^ 1;
      *(uint4*)(&ldsA[nxt][aRow * LDT + aCol])     = aR[0];
      *(uint4*)(&ldsA[nxt][aRow * LDT + aCol + 8]) = aR[1];
      *(uint4*)(&ldsB[nxt][tid * LDT])             = bR[0];
      *(uint4*)(&ldsB[nxt][tid * LDT + 8])         = bR[1];
      *(uint4*)(&ldsB[nxt][tid * LDT + 16])        = bR[2];
      *(uint4*)(&ldsB[nxt][tid * LDT + 24])        = bR[3];
    }
    __syncthreads();
  }

  // epilogue: C layout -> lanes 0-15 rows v, lanes 16-31 rows v+8, col = lane&15
  float bv[4];
#pragma unroll
  for (int ni = 0; ni < 4; ++ni)
    bv[ni] = bias[blockN + wn0 + ni * 16 + l15];
#pragma unroll
  for (int mi = 0; mi < 4; ++mi) {
#pragma unroll
    for (int ni = 0; ni < 4; ++ni) {
      const int col = blockN + wn0 + ni * 16 + l15;
#pragma unroll
      for (int v = 0; v < 8; ++v) {
        const int row = blockM + wm0 + mi * 16 + lhalf * 8 + v;
        out[(size_t)row * OUT_F + col] = c[mi][ni][v] + bv[ni];
      }
    }
  }
}

extern "C" void kernel_launch(void* const* d_in, const int* in_sizes, int n_in,
                              void* d_out, int out_size, void* d_ws, size_t ws_size,
                              hipStream_t stream) {
  (void)in_sizes; (void)n_in; (void)out_size; (void)ws_size;
  const float* x    = (const float*)d_in[0];
  const float* w    = (const float*)d_in[1];
  const float* bias = (const float*)d_in[2];
  const float* lA   = (const float*)d_in[3];
  const float* lB   = (const float*)d_in[4];
  float* out = (float*)d_out;

  // workspace: Xb (64 MB bf16) | Wb (32 MB bf16)
  unsigned short* Xb = (unsigned short*)d_ws;
  unsigned short* Wb = Xb + (size_t)M_TOK * IN_F;

  cvt_x_kernel<<<(M_TOK * IN_F) / (256 * 8), 256, 0, stream>>>(x, Xb);
  weff_kernel<<<OUT_F, 256, 0, stream>>>(w, lA, lB, Wb);
  lora_gemm_kernel<<<dim3(OUT_F / BN, M_TOK / BM), 256, 0, stream>>>(Xb, Wb, bias, out);
}
